// AttLayer_54683523613319
// MI455X (gfx1250) — matrix operand order repaired and checked by reading, not compiled, not run on hardware
//
#include <hip/hip_runtime.h>
#include <hip/hip_bf16.h>

// ---------- types ----------
typedef __bf16 bf16_t;
typedef __attribute__((ext_vector_type(16))) __bf16 v16bf;
typedef __attribute__((ext_vector_type(8)))  __bf16 v8bf;
typedef __attribute__((ext_vector_type(8)))  float  v8f;

static __device__ __forceinline__ bf16_t f32_to_bf16(float f) {
    // round-to-nearest-even, purely bitwise (no __bf16 arithmetic needed)
    unsigned u = __builtin_bit_cast(unsigned, f);
    u += 0x7FFFu + ((u >> 16) & 1u);
    unsigned short h = (unsigned short)(u >> 16);
    return __builtin_bit_cast(bf16_t, h);
}

// ---------- fp32 -> bf16 conversion ----------
__global__ void __launch_bounds__(256)
cvt_f32_to_bf16(const float* __restrict__ src, bf16_t* __restrict__ dst, size_t n) {
    size_t i = (size_t)blockIdx.x * blockDim.x + threadIdx.x;
    size_t stride = (size_t)gridDim.x * blockDim.x;
    for (; i < n; i += stride) dst[i] = f32_to_bf16(src[i]);
}

// ---------- WMMA fragment loaders (contiguous K per lane half, same order for A and B, wave32) ----------
// A: 16x32 bf16 tile from row-major matrix (row stride = ld elements).
// lanes 0-15 (m=lane):     halfs 0..15 -> K 0..15
// lanes 16-31 (m=lane-16): halfs 0..15 -> K 16..31
static __device__ __forceinline__ v16bf load_a_frag(const bf16_t* __restrict__ A,
                                                    int ld, int lane) {
    int row = lane & 15;
    int kb  = (lane >> 4) << 4;                 // 0 or 16
    const bf16_t* p = A + (size_t)row * ld + kb;
    v8bf lo = *(const v8bf*)(p);                // K kb..kb+7
    v8bf hi = *(const v8bf*)(p + 8);            // K kb+8..kb+15
    return __builtin_shufflevector(lo, hi, 0,1,2,3,4,5,6,7,8,9,10,11,12,13,14,15);
}

// B: 32x16 bf16 tile, sourced from an "N-major" buffer Bn where Bn[n][k] = B[k][n]
// (row stride = ld elements).  n = lane%16; lanes 0-15 hold K 0..15, lanes 16-31 K 16..31.
static __device__ __forceinline__ v16bf load_b_frag(const bf16_t* __restrict__ Bn,
                                                    int ld, int lane) {
    int n  = lane & 15;
    int kb = (lane >> 4) << 4;                  // 0 or 16
    return *(const v16bf*)(Bn + (size_t)n * ld + kb);   // 32B contiguous, aligned
}

#define WMMA_BF16(a, b, c) \
    __builtin_amdgcn_wmma_f32_16x16x32_bf16(false, (a), false, (b), (short)0, (c), false, false)

// Wave tile: 64(M) x 64(N) = 16 wmma accumulators.
// Block: 8 waves arranged 4(M) x 2(N) -> 256(M) x 128(N) block tile.
#define MI 4
#define NI 4

// Core K-loop shared by both GEMMs: accumulates a 64x64 wave tile.
static __device__ __forceinline__ void gemm_core(const bf16_t* __restrict__ A,
                                                 const bf16_t* __restrict__ Bn,
                                                 int m0, int n0, int K, int lane,
                                                 v8f acc[MI][NI]) {
    for (int k = 0; k < K; k += 32) {
        const bf16_t* Ak = A  + (size_t)m0 * K + k;
        const bf16_t* Bk = Bn + (size_t)n0 * K + k;
        if (k + 32 < K) {   // WGP-scope prefetch of the next K tile (global_prefetch_b8)
            __builtin_prefetch(Ak + 32 + (size_t)(lane & 15) * K, 0, 3);
            __builtin_prefetch(Bk + 32 + (size_t)(lane & 15) * K, 0, 3);
        }
        v16bf a[MI], b[NI];
        #pragma unroll
        for (int mi = 0; mi < MI; ++mi)
            a[mi] = load_a_frag(Ak + (size_t)(16 * mi) * K, K, lane);
        #pragma unroll
        for (int ni = 0; ni < NI; ++ni)
            b[ni] = load_b_frag(Bk + (size_t)(16 * ni) * K, K, lane);
        #pragma unroll
        for (int mi = 0; mi < MI; ++mi)
            #pragma unroll
            for (int ni = 0; ni < NI; ++ni)
                acc[mi][ni] = WMMA_BF16(a[mi], b[ni], acc[mi][ni]);
    }
}

// ---------- GEMM 1: qkv = text @ W^T + b, writes qkv (bf16) and qkvT (bf16, [B,D,S]) ----------
// X:  [M,K] bf16 row-major (M = B*S, K = D)
// Wn: [N,K] bf16 row-major (= W itself: W[n][k], N-major for the B operand)
__global__ void __launch_bounds__(256)
wmma_gemm_qkv(const bf16_t* __restrict__ X, const bf16_t* __restrict__ Wn,
              const float* __restrict__ bias,
              bf16_t* __restrict__ qkv, bf16_t* __restrict__ qkvT,
              int M, int N, int K, int S) {
    const int lane = threadIdx.x & 31;
    const int wave = threadIdx.x >> 5;          // 8 waves: 4 (M) x 2 (N)
    const int wm = wave & 3, wn = wave >> 2;
    const int m0 = blockIdx.y * 256 + wm * 64;
    const int n0 = blockIdx.x * 128 + wn * 64;

    v8f zero = {0.f,0.f,0.f,0.f,0.f,0.f,0.f,0.f};
    v8f acc[MI][NI];
    #pragma unroll
    for (int mi = 0; mi < MI; ++mi)
        #pragma unroll
        for (int ni = 0; ni < NI; ++ni) acc[mi][ni] = zero;

    gemm_core(X, Wn, m0, n0, K, lane, acc);

    const int mlo = (lane >> 4) << 3;           // 0 or 8 (C/D layout)
    const int nn  = lane & 15;
    #pragma unroll
    for (int mi = 0; mi < MI; ++mi)
      #pragma unroll
      for (int ni = 0; ni < NI; ++ni)
        #pragma unroll
        for (int r = 0; r < 8; ++r) {
            int m = m0 + mi * 16 + mlo + r;
            int n = n0 + ni * 16 + nn;
            bf16_t h = f32_to_bf16(acc[mi][ni][r] + bias[n]);
            qkv[(size_t)m * N + n] = h;
            int bidx = m / S, s = m - bidx * S; // transposed copy: qkvT[b][d][s]
            qkvT[((size_t)bidx * N + n) * S + s] = h;
        }
}

// ---------- GEMM 2/3: batched C = scale * (A @ Bn^T), f32 out ----------
// A:  [M,K] bf16 row-major (row stride K);  Bn: [N,K] bf16 row-major (row stride K)
__global__ void __launch_bounds__(256)
wmma_gemm_f32(const bf16_t* __restrict__ Abase, const bf16_t* __restrict__ Bbase,
              float* __restrict__ Cbase, int M, int N, int K,
              size_t sA, size_t sB, size_t sC, float scale) {
    const int lane = threadIdx.x & 31;
    const int wave = threadIdx.x >> 5;
    const int wm = wave & 3, wn = wave >> 2;
    const int m0 = blockIdx.y * 256 + wm * 64;
    const int n0 = blockIdx.x * 128 + wn * 64;
    const bf16_t* A  = Abase + (size_t)blockIdx.z * sA;
    const bf16_t* Bn = Bbase + (size_t)blockIdx.z * sB;
    float*        C  = Cbase + (size_t)blockIdx.z * sC;

    v8f zero = {0.f,0.f,0.f,0.f,0.f,0.f,0.f,0.f};
    v8f acc[MI][NI];
    #pragma unroll
    for (int mi = 0; mi < MI; ++mi)
        #pragma unroll
        for (int ni = 0; ni < NI; ++ni) acc[mi][ni] = zero;

    gemm_core(A, Bn, m0, n0, K, lane, acc);

    const int mlo = (lane >> 4) << 3;
    const int nn  = lane & 15;
    #pragma unroll
    for (int mi = 0; mi < MI; ++mi)
      #pragma unroll
      for (int ni = 0; ni < NI; ++ni)
        #pragma unroll
        for (int r = 0; r < 8; ++r) {
            int m = m0 + mi * 16 + mlo + r;
            int n = n0 + ni * 16 + nn;
            C[(size_t)m * N + n] = acc[mi][ni][r] * scale;
        }
}

// ---------- softmax across the batch dimension (torch legacy dim=0) ----------
#define BATCH 32
__global__ void __launch_bounds__(256)
softmax_over_batch(const float* __restrict__ scores, bf16_t* __restrict__ attn,
                   size_t plane) {
    size_t idx = (size_t)blockIdx.x * blockDim.x + threadIdx.x;
    if (idx >= plane) return;
    float v[BATCH];
    float mx = -3.402823466e38f;
    #pragma unroll
    for (int b = 0; b < BATCH; ++b) {
        v[b] = scores[(size_t)b * plane + idx];
        mx = fmaxf(mx, v[b]);
    }
    float sum = 0.f;
    #pragma unroll
    for (int b = 0; b < BATCH; ++b) { v[b] = __expf(v[b] - mx); sum += v[b]; }
    float inv = 1.f / sum;
    #pragma unroll
    for (int b = 0; b < BATCH; ++b)
        attn[(size_t)b * plane + idx] = f32_to_bf16(v[b] * inv);
}

// ---------- host launch ----------
extern "C" void kernel_launch(void* const* d_in, const int* in_sizes, int n_in,
                              void* d_out, int out_size, void* d_ws, size_t ws_size,
                              hipStream_t stream) {
    (void)in_sizes; (void)n_in; (void)out_size; (void)ws_size;

    const float* text = (const float*)d_in[0];   // [32,1024,1024]
    const float* W    = (const float*)d_in[1];   // [1024,1024]
    const float* bias = (const float*)d_in[2];   // [1024]
    float* out        = (float*)d_out;           // [32,1024,1024]

    constexpr int    Bb = 32, S = 1024, D = 1024;
    constexpr size_t MS = (size_t)Bb * S;        // 32768 rows
    constexpr size_t plane = (size_t)S * S;      // 1M

    // workspace carve-up (256B aligned)
    char* ws = (char*)d_ws;
    size_t off = 0;
    auto carve = [&](size_t bytes) -> void* {
        void* p = ws + off;
        off = (off + bytes + 255) & ~(size_t)255;
        return p;
    };
    bf16_t* text_bf = (bf16_t*)carve(MS * D * sizeof(bf16_t));             // 64 MB
    bf16_t* W_bf    = (bf16_t*)carve((size_t)D * D * sizeof(bf16_t));      //  2 MB
    bf16_t* qkv_bf  = (bf16_t*)carve(MS * D * sizeof(bf16_t));             // 64 MB
    bf16_t* qkvT_bf = (bf16_t*)carve(MS * D * sizeof(bf16_t));             // 64 MB
    float*  scores  = (float*) carve((size_t)Bb * plane * sizeof(float));  // 128 MB
    bf16_t* attn_bf = (bf16_t*)carve((size_t)Bb * plane * sizeof(bf16_t)); // 64 MB

    // 1) convert inputs to bf16
    cvt_f32_to_bf16<<<4096, 256, 0, stream>>>(text, text_bf, MS * D);
    cvt_f32_to_bf16<<<1024, 256, 0, stream>>>(W, W_bf, (size_t)D * D);

    // 2) qkv = text @ W^T + b   (also emits qkvT for GEMM3's B operand)
    dim3 g1(D / 128, (unsigned)(MS / 256));      // (8, 128)
    wmma_gemm_qkv<<<g1, 256, 0, stream>>>(text_bf, W_bf, bias, qkv_bf, qkvT_bf,
                                          (int)MS, D, D, S);

    // 3) scores[b] = (1/sqrt(D)) * qkv[b] @ qkv[b]^T
    dim3 g2(S / 128, S / 256, Bb);               // (8, 4, 32)
    wmma_gemm_f32<<<g2, 256, 0, stream>>>(qkv_bf, qkv_bf, scores, S, S, D,
                                          (size_t)S * D, (size_t)S * D, plane,
                                          1.0f / 32.0f);

    // 4) softmax over batch dim
    softmax_over_batch<<<(unsigned)((plane + 255) / 256), 256, 0, stream>>>(
        scores, attn_bf, plane);

    // 5) out[b] = attn[b] @ qkv[b]   (B operand from qkvT: Bn[n][k] = qkv[b][k][n])
    dim3 g3(D / 128, S / 256, Bb);
    wmma_gemm_f32<<<g3, 256, 0, stream>>>(attn_bf, qkvT_bf, out, S, D, S,
                                          plane, (size_t)S * D, (size_t)S * D,
                                          1.0f);
}
